// GeneratorDeConv_52252572123270
// MI455X (gfx1250) — compile-verified, matrix operand order checked by reading, not run on hardware
//
#include <hip/hip_runtime.h>
#include <hip/hip_bf16.h>

// ---------------------------------------------------------------------------
// Types / helpers
// ---------------------------------------------------------------------------
typedef __bf16 bf16_t;
typedef __attribute__((ext_vector_type(16))) __bf16 v16bf;
typedef __attribute__((ext_vector_type(8)))  __bf16 v8bf;
typedef __attribute__((ext_vector_type(8)))  float  v8f;

#define DEVFN __device__ __forceinline__

DEVFN float  bf2f(bf16_t h) { return (float)h; }
DEVFN bf16_t f2bf(float f)  { return (bf16_t)f; }

DEVFN v8f wmma_bf16(v16bf a, v16bf b, v8f c) {
  // D(16x16,f32) = A(16x32,bf16) x B(32x16,bf16) + C
  return __builtin_amdgcn_wmma_f32_16x16x32_bf16(
      /*neg_a=*/false, a, /*neg_b=*/false, b,
      /*c_mod=*/(short)0, c, /*reuse_a=*/false, /*reuse_b=*/false);
}

// ---------------------------------------------------------------------------
// Problem constants
// ---------------------------------------------------------------------------
constexpr int BATCH = 1024;
constexpr int L1_N  = 34 * 7 * 64;   // 15232 flattened (ho,wo,co) of layer-1 output

// d_out element offsets (reference tuple, concatenated flat, fp32)
constexpr size_t O0  = 0;                    // X[:,:,1:]        1024*30*5
constexpr size_t O1  = O0  + 153600;         // edge_index       2*870
constexpr size_t O2  = O1  + 1740;           // edge_attr[:,:,1:] 1024*870*3
constexpr size_t O3  = O2  + 2672640;        // 1-X[:,:,0]       1024*30
constexpr size_t O4  = O3  + 30720;          // 1-edge_attr[:,:,0] 1024*870
constexpr size_t O5  = O4  + 890880;         // batch            1024*30
constexpr size_t O6  = O5  + 30720;          // X_hard[:,:,1:]
constexpr size_t O7  = O6  + 153600;         // ea_hard[:,:,1:]
constexpr size_t O8  = O7  + 2672640;        // 1-X_hard[:,:,0]
constexpr size_t O9  = O8  + 30720;          // 1-ea_hard[:,:,0]
constexpr size_t O10 = O9  + 890880;         // log_prob (scalar)

// Workspace layout (bytes)
constexpr size_t wsalign(size_t x) { return (x + 255) & ~size_t(255); }
constexpr size_t WS_WE1   = 0;                                                  // 32 x 15232 bf16
constexpr size_t WS_W2    = wsalign(WS_WE1   + sizeof(bf16_t) * 32 * L1_N);
constexpr size_t WS_W3    = wsalign(WS_W2    + sizeof(bf16_t) * 3 * 3 * 64 * 32);
constexpr size_t WS_W4    = wsalign(WS_W3    + sizeof(bf16_t) * 3 * 3 * 32 * 32);
constexpr size_t WS_STATS = wsalign(WS_W4    + sizeof(bf16_t) * 3 * 5 * 32 * 1);
constexpr size_t WS_SCSH  = wsalign(WS_STATS + sizeof(float) * 256);
constexpr size_t WS_ACT1  = wsalign(WS_SCSH  + sizeof(float) * 256);
constexpr size_t WS_ACT2  = wsalign(WS_ACT1  + sizeof(bf16_t) * (size_t)BATCH * L1_N);
constexpr size_t WS_ACT3  = wsalign(WS_ACT2  + sizeof(bf16_t) * (size_t)BATCH * 36 * 27 * 32);
constexpr size_t WS_OUT   = wsalign(WS_ACT3  + sizeof(bf16_t) * (size_t)BATCH * 36 * 55 * 32);

// ---------------------------------------------------------------------------
// Init: zero BN stats + log_prob accumulator (deterministic per launch)
// ---------------------------------------------------------------------------
__global__ void k_init(float* __restrict__ stats, float* __restrict__ logprob) {
  int t = blockIdx.x * blockDim.x + threadIdx.x;
  if (t < 256) stats[t] = 0.0f;
  if (t == 0)  *logprob = 0.0f;
}

// ---------------------------------------------------------------------------
// Build expanded layer-1 weight:  We1[k][(ho*7+wo)*64+co] = W1[ho-k, wo, 0, co]
// so layer 1 becomes a dense GEMM  (B x 32) * (32 x 15232).
// ---------------------------------------------------------------------------
__global__ void k_prep_we1(const float* __restrict__ w1, bf16_t* __restrict__ we) {
  int tid = blockIdx.x * blockDim.x + threadIdx.x;
  if (tid >= 32 * L1_N) return;
  int k  = tid / L1_N;
  int j  = tid % L1_N;
  int ho = j / (7 * 64);
  int r  = j % (7 * 64);
  int wo = r / 64;
  int co = r % 64;
  int kh = ho - k;
  float v = (kh >= 0 && kh < 3) ? w1[(kh * 7 + wo) * 64 + co] : 0.0f;
  we[tid] = f2bf(v);
}

__global__ void k_prep_w(const float* __restrict__ w2, const float* __restrict__ w3,
                         const float* __restrict__ w4, bf16_t* __restrict__ w2b,
                         bf16_t* __restrict__ w3b, bf16_t* __restrict__ w4b) {
  int tid = blockIdx.x * blockDim.x + threadIdx.x;
  if (tid < 18432)              w2b[tid]          = f2bf(w2[tid]);
  else if (tid < 18432 + 9216)  w3b[tid - 18432]  = f2bf(w3[tid - 18432]);
  else if (tid < 18432 + 9216 + 480) w4b[tid - 27648] = f2bf(w4[tid - 27648]);
}

// ---------------------------------------------------------------------------
// Layer 1: dense GEMM  M=1024, K=32, N=15232 (one WMMA K-step per tile).
// Writes bf16 conv output + accumulates per-channel (co = col % 64) BN stats.
// Grid: (238 N-tiles of 64, 8 M-tiles of 128), block 256 (8 waves).
// ---------------------------------------------------------------------------
__global__ __launch_bounds__(256) void k_l1_gemm(const float* __restrict__ Z,
                                                 const bf16_t* __restrict__ We,
                                                 bf16_t* __restrict__ act1,
                                                 float* __restrict__ stats) {
  __shared__ __align__(64) bf16_t Al[128 * 32];
  __shared__ __align__(64) bf16_t Bt[64 * 32];     // transposed [n][k]
  __shared__ float ssum[64], ssq[64];

  const int t  = threadIdx.x;
  const int nb = blockIdx.x;          // 0..237
  const int b0 = blockIdx.y * 128;    // batch tile base
  if (t < 64) { ssum[t] = 0.0f; ssq[t] = 0.0f; }

  // Stage A (latent Z, fp32 -> bf16): 128x32
  for (int idx = t; idx < 128 * 32 / 4; idx += 256) {
    const float4 z4 = *(const float4*)(Z + (size_t)b0 * 32 + (size_t)idx * 4);
    bf16_t* dst = Al + idx * 4;
    dst[0] = f2bf(z4.x); dst[1] = f2bf(z4.y); dst[2] = f2bf(z4.z); dst[3] = f2bf(z4.w);
  }
  // Stage B transposed: Bt[n][k] = We[k][nb*64+n]  (vector read, scattered write)
  for (int idx = t; idx < 64 * 32 / 8; idx += 256) {
    const int k = idx / 8, n0 = (idx % 8) * 8;
    const v8bf v = *(const v8bf*)(We + (size_t)k * L1_N + nb * 64 + n0);
    #pragma unroll
    for (int e = 0; e < 8; e++) Bt[(n0 + e) * 32 + k] = v[e];
  }
  __syncthreads();

  const int wave = t >> 5, lane = t & 31;
  const int g = lane >> 4, ln = lane & 15;   // A: M=ln, K=g*16+e ; B: N=ln, K=g*16+e

  const v16bf a = *(const v16bf*)(Al + (wave * 16 + ln) * 32 + g * 16);
  v8f acc[4];
  #pragma unroll
  for (int i = 0; i < 4; i++) acc[i] = (v8f){0.f,0.f,0.f,0.f,0.f,0.f,0.f,0.f};

  #pragma unroll
  for (int n0 = 0; n0 < 4; n0++) {
    const v16bf b = *(const v16bf*)(Bt + (n0 * 16 + ln) * 32 + g * 16);
    acc[n0] = wmma_bf16(a, b, acc[n0]);
  }

  // Epilogue: store bf16 + per-channel stats (channel = column % 64)
  #pragma unroll
  for (int n0 = 0; n0 < 4; n0++) {
    float ls = 0.0f, lq = 0.0f;
    #pragma unroll
    for (int j = 0; j < 8; j++) {               // D layout: M=j+8*g, N=ln
      const int row = b0 + wave * 16 + j + 8 * g;
      const float x = acc[n0][j];
      act1[(size_t)row * L1_N + (size_t)nb * 64 + n0 * 16 + ln] = f2bf(x);
      ls += x; lq += x * x;
    }
    atomicAdd(&ssum[n0 * 16 + ln], ls);
    atomicAdd(&ssq [n0 * 16 + ln], lq);
  }
  __syncthreads();
  if (t < 64) { atomicAdd(stats + t, ssum[t]); atomicAdd(stats + 64 + t, ssq[t]); }
}

// ---------------------------------------------------------------------------
// BN finalize: per-channel affine fold   sc = g*rsqrt(var+eps), sh = b - mean*sc
// ---------------------------------------------------------------------------
__global__ void k_bn_fin(const float* __restrict__ stats, const float* __restrict__ gg,
                         const float* __restrict__ bb, float* __restrict__ sc,
                         float* __restrict__ sh, int C, float inv_cnt) {
  int c = threadIdx.x;
  if (c >= C) return;
  float mean = stats[c] * inv_cnt;
  float var  = stats[C + c] * inv_cnt - mean * mean;
  float rs   = rsqrtf(var + 1e-5f);
  sc[c] = gg[c] * rs;
  sh[c] = bb[c] - mean * gg[c] * rs;
}

// ---------------------------------------------------------------------------
// Implicit-GEMM gather deconv (layers 2 & 3).
// One workgroup = one output pixel (ho,wo) x 128-batch tile; 8 waves.
// Weight tensor staged into LDS *transposed* to [tap][co][ci] so every B
// fragment is one contiguous 32B ds_load.  A tile staged once per tap (full
// C_in) with fused BN+ReLU+bf16 convert via 16B vector global loads.
// ---------------------------------------------------------------------------
template<int HI, int WI, int CI, int HO, int WO, int CO,
         int KH, int KW, int SW, int PH, int PW, int NACC>
__global__ __launch_bounds__(256) void k_deconv(const bf16_t* __restrict__ ain,
                                                const float* __restrict__ sc,
                                                const float* __restrict__ sh,
                                                const bf16_t* __restrict__ wt,
                                                bf16_t* __restrict__ aout,
                                                float* __restrict__ stats) {
  __shared__ __align__(64) bf16_t Wt[KH * KW * CO * CI];  // [tap][co][ci]
  __shared__ __align__(64) bf16_t Al[128 * CI];
  __shared__ float scl[CI], shl[CI];
  __shared__ float ssum[CO], ssq[CO];

  const int t = threadIdx.x;
  __builtin_prefetch(wt, 0, 1);
  // Transposing weight stage: read [tap][ci][co] -> write [tap][co][ci]
  for (int idx = t; idx < KH * KW * CI * CO; idx += 256) {
    const int tap = idx / (CI * CO);
    const int r   = idx % (CI * CO);
    const int ci  = r / CO;
    const int co  = r % CO;
    Wt[(tap * CO + co) * CI + ci] = wt[idx];
  }
  for (int idx = t; idx < CI; idx += 256) { scl[idx] = sc[idx]; shl[idx] = sh[idx]; }
  if (t < CO) { ssum[t] = 0.0f; ssq[t] = 0.0f; }
  __syncthreads();

  const int pos = blockIdx.x;
  const int ho  = pos / WO, wo = pos % WO;
  const int b0  = blockIdx.y * 128;
  const int wave = t >> 5, lane = t & 31;
  const int g = lane >> 4, ln = lane & 15;

  v8f acc[NACC];
  #pragma unroll
  for (int i = 0; i < NACC; i++) acc[i] = (v8f){0.f,0.f,0.f,0.f,0.f,0.f,0.f,0.f};

  for (int kh = 0; kh < KH; kh++) {
    const int hi = ho + PH - kh;                       // stride_h == 1 everywhere
    if (hi < 0 || hi >= HI) continue;
    for (int kw = 0; kw < KW; kw++) {
      const int tw = wo + PW - kw;
      if (tw < 0 || (tw % SW) != 0) continue;
      const int wi = tw / SW;
      if (wi >= WI) continue;
      const int tap = kh * KW + kw;

      __syncthreads();
      // Stage A: 128 batch rows x CI channels, BN+ReLU fused, 16B vector loads
      for (int ch = t; ch < 128 * CI / 8; ch += 256) {
        const int r  = ch / (CI / 8);
        const int k0 = (ch % (CI / 8)) * 8;
        const v8bf v = *(const v8bf*)(ain + (((size_t)(b0 + r) * HI + hi) * WI + wi) * CI + k0);
        v8bf o;
        #pragma unroll
        for (int e = 0; e < 8; e++) {
          const float x = fmaxf(scl[k0 + e] * bf2f(v[e]) + shl[k0 + e], 0.0f);
          o[e] = f2bf(x);
        }
        *(v8bf*)(Al + r * CI + k0) = o;
      }
      __syncthreads();

      #pragma unroll
      for (int c0 = 0; c0 < CI; c0 += 32) {
        const v16bf a = *(const v16bf*)(Al + (wave * 16 + ln) * CI + c0 + g * 16);
        #pragma unroll
        for (int n0 = 0; n0 < NACC; n0++) {
          const v16bf b = *(const v16bf*)(Wt + ((size_t)tap * CO + n0 * 16 + ln) * CI + c0 + g * 16);
          acc[n0] = wmma_bf16(a, b, acc[n0]);
        }
      }
    }
  }

  // Epilogue: bf16 store + channel stats
  const size_t spat = (size_t)HO * WO;
  #pragma unroll
  for (int n0 = 0; n0 < NACC; n0++) {
    float ls = 0.0f, lq = 0.0f;
    #pragma unroll
    for (int j = 0; j < 8; j++) {
      const int row = b0 + wave * 16 + j + 8 * g;
      const float x = acc[n0][j];
      aout[((size_t)row * spat + pos) * CO + n0 * 16 + ln] = f2bf(x);
      ls += x; lq += x * x;
    }
    atomicAdd(&ssum[n0 * 16 + ln], ls);
    atomicAdd(&ssq [n0 * 16 + ln], lq);
  }
  __syncthreads();
  if (t < CO) { atomicAdd(stats + t, ssum[t]); atomicAdd(stats + CO + t, ssq[t]); }
}

// ---------------------------------------------------------------------------
// Layer 4: C_out = 1 (N padded to 16, only column 0 stored), fp32 output.
// ---------------------------------------------------------------------------
__global__ __launch_bounds__(256) void k_l4(const bf16_t* __restrict__ ain,
                                            const float* __restrict__ sc,
                                            const float* __restrict__ sh,
                                            const bf16_t* __restrict__ wt,
                                            float* __restrict__ out126) {
  constexpr int HI = 36, WI = 55, CI = 32, WO = 105, KH = 3, KW = 5, PH = 1, PW = 4;
  __shared__ __align__(64) bf16_t Wl[KH * KW * CI];
  __shared__ __align__(64) bf16_t Al[128 * 32];
  __shared__ float scl[CI], shl[CI];

  const int t = threadIdx.x;
  for (int idx = t; idx < KH * KW * CI; idx += 256) Wl[idx] = wt[idx];
  for (int idx = t; idx < CI; idx += 256) { scl[idx] = sc[idx]; shl[idx] = sh[idx]; }

  const int pos = blockIdx.x;
  const int ho  = pos / WO, wo = pos % WO;
  const int b0  = blockIdx.y * 128;
  const int wave = t >> 5, lane = t & 31;
  const int g = lane >> 4, ln = lane & 15;

  v8f acc = (v8f){0.f,0.f,0.f,0.f,0.f,0.f,0.f,0.f};

  for (int kh = 0; kh < KH; kh++) {
    const int hi = ho + PH - kh;
    if (hi < 0 || hi >= HI) continue;
    for (int kw = 0; kw < KW; kw++) {
      const int tw = wo + PW - kw;
      if (tw < 0 || (tw & 1)) continue;
      const int wi = tw >> 1;
      if (wi >= WI) continue;

      __syncthreads();
      for (int ch = t; ch < 128 * 32 / 8; ch += 256) {
        const int r  = ch >> 2;
        const int k0 = (ch & 3) * 8;
        const v8bf v = *(const v8bf*)(ain + (((size_t)(b0 + r) * HI + hi) * WI + wi) * CI + k0);
        v8bf o;
        #pragma unroll
        for (int e = 0; e < 8; e++) {
          const float x = fmaxf(scl[k0 + e] * bf2f(v[e]) + shl[k0 + e], 0.0f);
          o[e] = f2bf(x);
        }
        *(v8bf*)(Al + r * 32 + k0) = o;
      }
      __syncthreads();

      const v16bf a = *(const v16bf*)(Al + (wave * 16 + ln) * 32 + g * 16);
      v16bf b;
      if (ln == 0) {
        b = *(const v16bf*)(Wl + (kh * KW + kw) * CI + g * 16);   // contiguous K
      } else {
        #pragma unroll
        for (int e = 0; e < 16; e++) b[e] = f2bf(0.0f);
      }
      acc = wmma_bf16(a, b, acc);
    }
  }

  if (ln == 0) {
    #pragma unroll
    for (int j = 0; j < 8; j++) {
      const int row = b0 + wave * 16 + j + 8 * g;
      out126[(size_t)row * 3780 + pos] = acc[j];
    }
  }
}

// ---------------------------------------------------------------------------
// Heads
// ---------------------------------------------------------------------------
__global__ void k_node_head(const float* __restrict__ o, float* __restrict__ out) {
  __shared__ float red[256];
  const int tid = blockIdx.x * blockDim.x + threadIdx.x;
  float lp = 0.0f;
  if (tid < BATCH * 30) {
    const int b = tid / 30, n = tid % 30;
    const float* l = o + (size_t)b * 3780 + n * 126;
    float m = l[0];
    #pragma unroll
    for (int i = 1; i < 6; i++) m = fmaxf(m, l[i]);
    float e[6], s = 0.0f;
    #pragma unroll
    for (int i = 0; i < 6; i++) { e[i] = expf(l[i] - m); s += e[i]; }
    const float inv = 1.0f / s;
    int am = 0; float bm = e[0];
    #pragma unroll
    for (int i = 1; i < 6; i++) if (e[i] > bm) { bm = e[i]; am = i; }
    #pragma unroll
    for (int i = 1; i < 6; i++) {
      out[O0 + (size_t)tid * 5 + (i - 1)] = e[i] * inv;
      out[O6 + (size_t)tid * 5 + (i - 1)] = (i == am) ? 1.0f : 0.0f;
    }
    out[O3 + tid] = 1.0f - e[0] * inv;
    out[O8 + tid] = (am == 0) ? 0.0f : 1.0f;
    lp = logf(e[am] * inv + 1e-7f);
  }
  red[threadIdx.x] = lp;
  __syncthreads();
  for (int s2 = 128; s2 > 0; s2 >>= 1) {
    if (threadIdx.x < s2) red[threadIdx.x] += red[threadIdx.x + s2];
    __syncthreads();
  }
  if (threadIdx.x == 0) atomicAdd(out + O10, red[0]);
}

__global__ void k_edge_head(const float* __restrict__ o, float* __restrict__ out) {
  __shared__ float red[256];
  const int tid = blockIdx.x * blockDim.x + threadIdx.x;
  float lp = 0.0f;
  if (tid < BATCH * 435) {
    const int b = tid / 435, e = tid % 435;
    int i = 0, off = 0;
    while (off + (29 - i) <= e) { off += 29 - i; ++i; }
    const int j = i + 1 + (e - off);
    const float* l = o + (size_t)b * 3780 + i * 126 + 6 + j * 4;
    float m = fmaxf(fmaxf(l[0], l[1]), fmaxf(l[2], l[3]));
    float ex[4], s = 0.0f;
    #pragma unroll
    for (int k = 0; k < 4; k++) { ex[k] = expf(l[k] - m); s += ex[k]; }
    const float inv = 1.0f / s;
    int am = 0; float bm = ex[0];
    #pragma unroll
    for (int k = 1; k < 4; k++) if (ex[k] > bm) { bm = ex[k]; am = k; }
    const size_t r1 = (size_t)b * 870 + e, r2 = r1 + 435;
    #pragma unroll
    for (int k = 1; k < 4; k++) {
      const float p = ex[k] * inv;
      const float h = (k == am) ? 1.0f : 0.0f;
      out[O2 + r1 * 3 + (k - 1)] = p;  out[O2 + r2 * 3 + (k - 1)] = p;
      out[O7 + r1 * 3 + (k - 1)] = h;  out[O7 + r2 * 3 + (k - 1)] = h;
    }
    const float v0 = 1.0f - ex[0] * inv;
    const float h0 = (am == 0) ? 0.0f : 1.0f;
    out[O4 + r1] = v0; out[O4 + r2] = v0;
    out[O9 + r1] = h0; out[O9 + r2] = h0;
    lp = logf(ex[am] * inv + 1e-7f);
  }
  red[threadIdx.x] = lp;
  __syncthreads();
  for (int s2 = 128; s2 > 0; s2 >>= 1) {
    if (threadIdx.x < s2) red[threadIdx.x] += red[threadIdx.x + s2];
    __syncthreads();
  }
  if (threadIdx.x == 0) atomicAdd(out + O10, red[0]);
}

__global__ void k_static(float* __restrict__ out) {
  const int tid = blockIdx.x * blockDim.x + threadIdx.x;
  if (tid < 435) {
    int i = 0, off = 0;
    while (off + (29 - i) <= tid) { off += 29 - i; ++i; }
    const int j = i + 1 + (tid - off);
    out[O1 + tid]        = (float)i;
    out[O1 + 435 + tid]  = (float)j;
    out[O1 + 870 + tid]  = (float)j;
    out[O1 + 1305 + tid] = (float)i;
  }
  const int k = tid - 435;
  if (k >= 0 && k < BATCH * 30) out[O5 + k] = (float)(k / 30);
}

// ---------------------------------------------------------------------------
// Launch
// ---------------------------------------------------------------------------
extern "C" void kernel_launch(void* const* d_in, const int* in_sizes, int n_in,
                              void* d_out, int out_size, void* d_ws, size_t ws_size,
                              hipStream_t stream) {
  (void)in_sizes; (void)n_in; (void)out_size; (void)ws_size;

  const float* Z  = (const float*)d_in[0];
  const float* W1 = (const float*)d_in[1];
  const float* W2 = (const float*)d_in[2];
  const float* W3 = (const float*)d_in[3];
  const float* W4 = (const float*)d_in[4];
  const float* g1 = (const float*)d_in[5];
  const float* b1 = (const float*)d_in[6];
  const float* g2 = (const float*)d_in[7];
  const float* b2 = (const float*)d_in[8];
  const float* g3 = (const float*)d_in[9];
  const float* b3 = (const float*)d_in[10];

  char* ws = (char*)d_ws;
  bf16_t* we1  = (bf16_t*)(ws + WS_WE1);
  bf16_t* w2b  = (bf16_t*)(ws + WS_W2);
  bf16_t* w3b  = (bf16_t*)(ws + WS_W3);
  bf16_t* w4b  = (bf16_t*)(ws + WS_W4);
  float*  stats= (float*)(ws + WS_STATS);   // [sum1(64) sq1(64) sum2(32) sq2(32) sum3(32) sq3(32)]
  float*  scsh = (float*)(ws + WS_SCSH);    // [sc1(64) sh1(64) sc2(32) sh2(32) sc3(32) sh3(32)]
  bf16_t* act1 = (bf16_t*)(ws + WS_ACT1);
  bf16_t* act2 = (bf16_t*)(ws + WS_ACT2);
  bf16_t* act3 = (bf16_t*)(ws + WS_ACT3);
  float*  o126 = (float*)(ws + WS_OUT);
  float*  out  = (float*)d_out;

  // 0) zero stats + log_prob
  k_init<<<1, 256, 0, stream>>>(stats, out + O10);

  // 1) weight prep
  k_prep_we1<<<(32 * L1_N + 255) / 256, 256, 0, stream>>>(W1, we1);
  k_prep_w<<<(18432 + 9216 + 480 + 255) / 256, 256, 0, stream>>>(W2, W3, W4, w2b, w3b, w4b);

  // 2) layer 1 GEMM (M=1024,K=32,N=15232) + stats
  k_l1_gemm<<<dim3(L1_N / 64, BATCH / 128), 256, 0, stream>>>(Z, we1, act1, stats);
  k_bn_fin<<<1, 64, 0, stream>>>(stats, g1, b1, scsh, scsh + 64, 64,
                                 1.0f / (float)(BATCH * 34 * 7));

  // 3) layer 2: (B,34,7,64)->(B,36,27,32), stride_w=4
  k_deconv<34, 7, 64, 36, 27, 32, 3, 3, 4, 0, 0, 2>
      <<<dim3(36 * 27, BATCH / 128), 256, 0, stream>>>(act1, scsh, scsh + 64, w2b,
                                                       act2, stats + 128);
  k_bn_fin<<<1, 32, 0, stream>>>(stats + 128, g2, b2, scsh + 128, scsh + 160, 32,
                                 1.0f / (float)(BATCH * 36 * 27));

  // 4) layer 3: (B,36,27,32)->(B,36,55,32), stride_w=2, pad_h=1
  k_deconv<36, 27, 32, 36, 55, 32, 3, 3, 2, 1, 0, 2>
      <<<dim3(36 * 55, BATCH / 128), 256, 0, stream>>>(act2, scsh + 128, scsh + 160, w3b,
                                                       act3, stats + 192);
  k_bn_fin<<<1, 32, 0, stream>>>(stats + 192, g3, b3, scsh + 192, scsh + 224, 32,
                                 1.0f / (float)(BATCH * 36 * 55));

  // 5) layer 4: (B,36,55,32)->(B,36,105,1) fp32 logits
  k_l4<<<dim3(36 * 105, BATCH / 128), 256, 0, stream>>>(act3, scsh + 192, scsh + 224,
                                                        w4b, o126);

  // 6) heads
  k_node_head<<<(BATCH * 30 + 255) / 256, 256, 0, stream>>>(o126, out);
  k_edge_head<<<(BATCH * 435 + 255) / 256, 256, 0, stream>>>(o126, out);
  k_static<<<(435 + BATCH * 30 + 255) / 256, 256, 0, stream>>>(out);
}
